// GLABlock_89507118448800
// MI455X (gfx1250) — matrix operand order repaired and checked by reading, not compiled, not run on hardware
//
#include <hip/hip_runtime.h>

typedef __attribute__((ext_vector_type(16))) __bf16 v16bf;
typedef __attribute__((ext_vector_type(8)))  float  v8f;

// ---- problem dimensions (from reference setup_inputs) ----
#define NB     4
#define TSEQ   2048
#define BT     8192          // NB*TSEQ
#define HS     1024
#define DK     512
#define DV     1024
#define NH     4
#define DKH    128           // DK/NH
#define DVH    256           // DV/NH
#define DVHH   128           // half of DVH (per-workgroup split)
#define ISZ    2816
#define CHUNKC 64
#define NCHUNK 32            // TSEQ/CHUNKC

// ---------------- bf16 helpers ----------------
__device__ __forceinline__ unsigned short f2bf(float x) {
  union { float f; unsigned int u; } c; c.f = x;
  unsigned int u = c.u;
  return (unsigned short)((u + 0x7FFFu + ((u >> 16) & 1u)) >> 16);
}

union FragBF { v16bf v; unsigned short u[16]; uint4 q[2]; };

// A-matrix 16x32 bf16 fragment from row-major (K-contiguous) source.
// Layout: lane m = lane&15; lanes 0-15 hold K k0+[0..15],
// lanes 16-31 hold K k0+[16..31]  -> two aligned 16B loads.
__device__ __forceinline__ v16bf load_a_frag(const unsigned short* A, int lda,
                                             int row0, int k0, int lane) {
  int m = row0 + (lane & 15);
  int koff = (lane >> 4) << 4;
  FragBF f;
  const uint4* p = (const uint4*)(A + (size_t)m * lda + k0 + koff);
  f.q[0] = p[0];
  f.q[1] = p[1];              // +8 elements
  return f.v;
}

// B-matrix 32x16 bf16 fragment from a TRANSPOSED (N-major, K-contiguous)
// source: B[k][n] = Bt[n][k]. Lanes 0-15: col n, K k0+[0..15];
// lanes 16-31: K k0+[16..31] -> two aligned 16B loads.
__device__ __forceinline__ v16bf load_bt_frag(const unsigned short* Bt, int ldb,
                                              int k0, int col0, int lane) {
  int n = col0 + (lane & 15);
  int kb = k0 + ((lane >> 4) << 4);
  FragBF f;
  const uint4* p = (const uint4*)(Bt + (size_t)n * ldb + kb);
  f.q[0] = p[0];
  f.q[1] = p[1];
  return f.v;
}

// Same, but the transposed source is f32 (GLA state S^T) -> 4x float4 + cvt.
__device__ __forceinline__ v16bf load_bt_frag_f32(const float* St, int ldb,
                                                  int k0, int col0, int lane) {
  int n = col0 + (lane & 15);
  int kb = k0 + ((lane >> 4) << 4);
  FragBF f;
  const float4* p = (const float4*)(St + (size_t)n * ldb + kb);
#pragma unroll
  for (int j = 0; j < 4; ++j) {
    float4 t = p[j];
    f.u[4 * j + 0] = f2bf(t.x);
    f.u[4 * j + 1] = f2bf(t.y);
    f.u[4 * j + 2] = f2bf(t.z);
    f.u[4 * j + 3] = f2bf(t.w);
  }
  return f.v;
}

// ---------------- block reduction ----------------
__device__ __forceinline__ float block_reduce_sum(float val, int tid) {
  __shared__ float red[8];
  __shared__ float resbuf;
#pragma unroll
  for (int off = 16; off > 0; off >>= 1) val += __shfl_down(val, off, 32);
  if ((tid & 31) == 0) red[tid >> 5] = val;
  __syncthreads();
  float t = (tid < 8) ? red[tid] : 0.f;
#pragma unroll
  for (int off = 4; off > 0; off >>= 1) t += __shfl_down(t, off, 32);
  if (tid == 0) resbuf = t;
  __syncthreads();
  return resbuf;
}

// ---------------- elementwise / norm kernels ----------------
// convert + transpose: src f32 [K][N] -> dst bf16 [N][K]
__global__ __launch_bounds__(256) void k_cvt_t(const float* __restrict__ s,
                                               unsigned short* __restrict__ d,
                                               int K, int N) {
  long i = (long)blockIdx.x * 256 + threadIdx.x;
  if (i >= (long)K * N) return;
  int kk = (int)(i / N), n = (int)(i % N);
  d[(size_t)n * K + kk] = f2bf(s[i]);
}

__global__ __launch_bounds__(256) void k_rmsnorm_bf16(const float* __restrict__ x,
                                                      const float* __restrict__ w,
                                                      unsigned short* __restrict__ hb, int D) {
  long row = blockIdx.x;
  const float* xr = x + row * D;
  float ss = 0.f;
  for (int i = threadIdx.x; i < D; i += 256) { float t = xr[i]; ss += t * t; }
  float tot = block_reduce_sum(ss, threadIdx.x);
  float rinv = rsqrtf(tot / (float)D + 1e-6f);
  for (int i = threadIdx.x; i < D; i += 256) hb[row * D + i] = f2bf(xr[i] * rinv * w[i]);
}

__global__ __launch_bounds__(256) void k_resnorm(const float* __restrict__ a,
                                                 const float* __restrict__ b,
                                                 const float* __restrict__ w,
                                                 float* __restrict__ x2,
                                                 unsigned short* __restrict__ nb, int D) {
  long row = blockIdx.x;
  float ss = 0.f;
  for (int i = threadIdx.x; i < D; i += 256) {
    float t = a[row * D + i] + b[row * D + i];
    x2[row * D + i] = t;
    ss += t * t;
  }
  float tot = block_reduce_sum(ss, threadIdx.x);
  float rinv = rsqrtf(tot / (float)D + 1e-6f);
  for (int i = threadIdx.x; i < D; i += 256)
    nb[row * D + i] = f2bf(x2[row * D + i] * rinv * w[i]);
}

// o = rms(o over dv) * gnorm_w * silu(g_out), per (row, head); dv = 256 = blockDim
__global__ __launch_bounds__(256) void k_gatenorm(const float* __restrict__ o,
                                                  const float* __restrict__ gout,
                                                  const float* __restrict__ gw,
                                                  unsigned short* __restrict__ dst) {
  long rh = blockIdx.x;
  long row = rh >> 2;
  int head = (int)(rh & 3);
  long base = row * DV + (long)head * DVH;
  int e = threadIdx.x;
  float val = o[base + e];
  float tot = block_reduce_sum(val * val, threadIdx.x);
  float rinv = rsqrtf(tot / (float)DVH + 1e-6f);
  float g = gout[base + e];
  float sig = 1.f / (1.f + __expf(-g));
  dst[base + e] = f2bf(val * rinv * gw[e] * (g * sig));
}

// glog = log_sigmoid(z + bgk2)/16, in place on z [BT, DK]
__global__ __launch_bounds__(256) void k_glog(float* __restrict__ z,
                                              const float* __restrict__ bgk2, long n) {
  long i = (long)blockIdx.x * 256 + threadIdx.x;
  if (i >= n) return;
  float t = z[i] + bgk2[i % DK];
  float ls = fminf(t, 0.f) - log1pf(__expf(-fabsf(t)));
  z[i] = ls * 0.0625f;   // / GATE_LOGIT_NORMALIZER
}

// W' = Wgk1(1024x16) @ Wgk2(16x512), stored transposed bf16 [DK][HS]
__global__ __launch_bounds__(256) void k_wgk12(const float* __restrict__ w1,
                                               const float* __restrict__ w2,
                                               unsigned short* __restrict__ dstT) {
  int idx = blockIdx.x * 256 + threadIdx.x;
  if (idx >= HS * DK) return;
  int i = idx / DK, j = idx % DK;
  float s = 0.f;
#pragma unroll
  for (int r = 0; r < 16; ++r) s += w1[i * 16 + r] * w2[r * DK + j];
  dstT[(size_t)j * HS + i] = f2bf(s);
}

__global__ __launch_bounds__(256) void k_silu_mul(const float* __restrict__ g,
                                                  const float* __restrict__ u,
                                                  unsigned short* __restrict__ d, long n) {
  long i = (long)blockIdx.x * 256 + threadIdx.x;
  if (i >= n) return;
  float gv = g[i];
  d[i] = f2bf((gv / (1.f + __expf(-gv))) * u[i]);
}

__global__ __launch_bounds__(256) void k_final_add(const float* __restrict__ a,
                                                   const float* __restrict__ b,
                                                   float* __restrict__ out, long n) {
  long i = (long)blockIdx.x * 256 + threadIdx.x;
  if (i < n) out[i] = a[i] + b[i];
}

// ------- generic bf16 WMMA GEMM: C[M,N] f32 = A[M,K]bf16 @ Bt[N,K]bf16^T -------
// One wave computes a 16x64 stripe (4 WMMA tiles sharing one A fragment).
__global__ __launch_bounds__(256) void k_gemm_bf16(const unsigned short* __restrict__ A,
                                                   const unsigned short* __restrict__ Bt,
                                                   float* __restrict__ C,
                                                   int M, int N, int K) {
  int lane = threadIdx.x & 31;
  int wave = threadIdx.x >> 5;
  int ntn = N >> 6;                       // 64-wide N stripes
  long tile = (long)blockIdx.x * 8 + wave;
  long total = (long)(M >> 4) * ntn;
  if (tile >= total) return;              // wave-uniform: EXEC stays all-ones
  int tm = (int)(tile / ntn);
  int tn = (int)(tile % ntn);
  int row0 = tm << 4, col0 = tn << 6;
  v8f acc0 = {}, acc1 = {}, acc2 = {}, acc3 = {};
  for (int kk = 0; kk < K; kk += 32) {
    __builtin_prefetch(A + (size_t)row0 * K + kk + 128, 0, 0);
    v16bf a  = load_a_frag(A, K, row0, kk, lane);
    v16bf b0 = load_bt_frag(Bt, K, kk, col0, lane);
    v16bf b1 = load_bt_frag(Bt, K, kk, col0 + 16, lane);
    v16bf b2 = load_bt_frag(Bt, K, kk, col0 + 32, lane);
    v16bf b3 = load_bt_frag(Bt, K, kk, col0 + 48, lane);
    acc0 = __builtin_amdgcn_wmma_f32_16x16x32_bf16(false, a, false, b0, (short)0, acc0, false, false);
    acc1 = __builtin_amdgcn_wmma_f32_16x16x32_bf16(false, a, false, b1, (short)0, acc1, false, false);
    acc2 = __builtin_amdgcn_wmma_f32_16x16x32_bf16(false, a, false, b2, (short)0, acc2, false, false);
    acc3 = __builtin_amdgcn_wmma_f32_16x16x32_bf16(false, a, false, b3, (short)0, acc3, false, false);
  }
  int n = lane & 15;
  int mb = row0 + ((lane >> 4) << 3);
#pragma unroll
  for (int r = 0; r < 8; ++r) {
    size_t rowoff = (size_t)(mb + r) * N + col0 + n;
    C[rowoff]      = acc0[r];
    C[rowoff + 16] = acc1[r];
    C[rowoff + 32] = acc2[r];
    C[rowoff + 48] = acc3[r];
  }
}

// ---------------- chunked GLA attention ----------------
// grid = NB*NH*2 workgroups: each handles (batch, head, half-of-dv) and scans
// the 32 chunks sequentially with the 128x128 f32 state S^T resident in LDS.
__global__ __launch_bounds__(256) void k_gla_attn(const float* __restrict__ q,
                                                  const float* __restrict__ k,
                                                  const float* __restrict__ v,
                                                  const float* __restrict__ g,
                                                  float* __restrict__ o) {
  __shared__ __align__(16) float          ST[DVHH * DKH];      // 64 KB  S^T: [e][d]
  __shared__ __align__(16) float          bq[CHUNKC * DKH];    // 32 KB  cumsum(glog) [r][d]
  __shared__               float          blast[DKH];
  __shared__ __align__(16) unsigned short qe[CHUNKC * DKH];    // 16 KB  [r][d]
  __shared__ __align__(16) unsigned short ktb[CHUNKC * DKH];   // 16 KB  ke [r][d], then kd^T [d][r]
  __shared__ __align__(16) unsigned short vtT[DVHH * CHUNKC];  // 16 KB  v^T [e][r]
  __shared__ __align__(16) unsigned short Abf[CHUNKC * CHUNKC];//  8 KB  [c][s]

  const int tid  = threadIdx.x;
  const int lane = tid & 31;
  const int wave = tid >> 5;
  const int half = blockIdx.x & 1;
  const int bh   = blockIdx.x >> 1;
  const int hh   = bh & (NH - 1);
  const int bb   = bh >> 2;                  // NH == 4
  const long rowbase = (long)bb * TSEQ;
  const int qcol = hh * DKH;
  const int vcol = hh * DVH + half * DVHH;
  const float scale = 0.08838834764831845f;  // dk^-0.5, dk=128

  for (int i = tid; i < DVHH * DKH; i += 256) ST[i] = 0.f;
  __syncthreads();

  for (int c = 0; c < NCHUNK; ++c) {
    const long trow = rowbase + (long)c * CHUNKC;

    // phase 1: per-column cumsum of gate logits within the chunk
    if (tid < DKH) {
      float s = 0.f;
      for (int r = 0; r < CHUNKC; ++r) {
        s += g[(trow + r) * DK + qcol + tid];
        bq[r * DKH + tid] = s;
      }
      blast[tid] = s;
    }
    __syncthreads();

    // phase 2: qe = q*exp(b)*scale ; ke = k*exp(-b)  (float4 loads)
    for (int i = tid; i < (CHUNKC * DKH) / 4; i += 256) {
      int r  = i >> 5;                 // DKH/4 == 32
      int c4 = (i & 31) << 2;
      const float4 qv = *(const float4*)&q[(trow + r) * DK + qcol + c4];
      const float4 kv = *(const float4*)&k[(trow + r) * DK + qcol + c4];
      const float4 bv = *(const float4*)&bq[r * DKH + c4];
      union { unsigned short s[4]; uint2 u; } pq, pk;
      pq.s[0] = f2bf(qv.x * __expf(bv.x) * scale);
      pq.s[1] = f2bf(qv.y * __expf(bv.y) * scale);
      pq.s[2] = f2bf(qv.z * __expf(bv.z) * scale);
      pq.s[3] = f2bf(qv.w * __expf(bv.w) * scale);
      pk.s[0] = f2bf(kv.x * __expf(-bv.x));
      pk.s[1] = f2bf(kv.y * __expf(-bv.y));
      pk.s[2] = f2bf(kv.z * __expf(-bv.z));
      pk.s[3] = f2bf(kv.w * __expf(-bv.w));
      *(uint2*)&qe[r * DKH + c4]  = pq.u;
      *(uint2*)&ktb[r * DKH + c4] = pk.u;
    }
    __syncthreads();

    // phase 3: A = qe @ ke^T (64x64), tril mask; 16 tiles -> 2 per wave
#pragma unroll
    for (int i = 0; i < 2; ++i) {
      int tile = wave * 2 + i;
      int tm = tile >> 2, tn = tile & 3;
      v8f acc = {};
#pragma unroll
      for (int kk = 0; kk < DKH; kk += 32) {
        v16bf a = load_a_frag(qe, DKH, tm * 16, kk, lane);
        v16bf b = load_bt_frag(ktb, DKH, kk, tn * 16, lane);
        acc = __builtin_amdgcn_wmma_f32_16x16x32_bf16(false, a, false, b, (short)0, acc, false, false);
      }
      int n = tn * 16 + (lane & 15);
      int mb = tm * 16 + ((lane >> 4) << 3);
#pragma unroll
      for (int r = 0; r < 8; ++r) {
        int m = mb + r;
        Abf[m * CHUNKC + n] = (m >= n) ? f2bf(acc[r]) : (unsigned short)0;
      }
    }
    __syncthreads();

    // phase 4: kd^T = (k*exp(blast-b))^T ; v^T  (float4 global loads)
    for (int i = tid; i < (CHUNKC * DKH) / 4; i += 256) {
      int r  = i >> 5;
      int c4 = (i & 31) << 2;
      const float4 kv = *(const float4*)&k[(trow + r) * DK + qcol + c4];
      const float4 bv = *(const float4*)&bq[r * DKH + c4];
      ktb[(c4 + 0) * CHUNKC + r] = f2bf(kv.x * __expf(blast[c4 + 0] - bv.x));
      ktb[(c4 + 1) * CHUNKC + r] = f2bf(kv.y * __expf(blast[c4 + 1] - bv.y));
      ktb[(c4 + 2) * CHUNKC + r] = f2bf(kv.z * __expf(blast[c4 + 2] - bv.z));
      ktb[(c4 + 3) * CHUNKC + r] = f2bf(kv.w * __expf(blast[c4 + 3] - bv.w));
    }
    for (int i = tid; i < (CHUNKC * DVHH) / 4; i += 256) {
      int r  = i >> 5;
      int e4 = (i & 31) << 2;
      const float4 vv = *(const float4*)&v[(trow + r) * DV + vcol + e4];
      vtT[(e4 + 0) * CHUNKC + r] = f2bf(vv.x);
      vtT[(e4 + 1) * CHUNKC + r] = f2bf(vv.y);
      vtT[(e4 + 2) * CHUNKC + r] = f2bf(vv.z);
      vtT[(e4 + 3) * CHUNKC + r] = f2bf(vv.w);
    }
    __syncthreads();

    // phase 5: o = A@v (intra) + qe@S (inter); 32 tiles -> 4 per wave
#pragma unroll
    for (int i = 0; i < 4; ++i) {
      int tile = wave * 4 + i;
      int tm = tile >> 3, tn = tile & 7;
      v8f acc = {};
#pragma unroll
      for (int kk = 0; kk < CHUNKC; kk += 32) {
        v16bf a = load_a_frag(Abf, CHUNKC, tm * 16, kk, lane);
        v16bf b = load_bt_frag(vtT, CHUNKC, kk, tn * 16, lane);
        acc = __builtin_amdgcn_wmma_f32_16x16x32_bf16(false, a, false, b, (short)0, acc, false, false);
      }
#pragma unroll
      for (int kk = 0; kk < DKH; kk += 32) {
        v16bf a = load_a_frag(qe, DKH, tm * 16, kk, lane);
        v16bf b = load_bt_frag_f32(ST, DKH, kk, tn * 16, lane);
        acc = __builtin_amdgcn_wmma_f32_16x16x32_bf16(false, a, false, b, (short)0, acc, false, false);
      }
      int n = tn * 16 + (lane & 15);
      int mb = tm * 16 + ((lane >> 4) << 3);
#pragma unroll
      for (int r = 0; r < 8; ++r)
        o[(trow + mb + r) * DV + vcol + n] = acc[r];
    }
    __syncthreads();   // all reads of S^T done before the update

    // phase 6: S^T[e][d] = S^T[e][d]*exp(blast[d]) + (kd^T @ v)[d][e]
    // 64 tiles (d-tiles x e-tiles) -> 8 per wave
#pragma unroll
    for (int i = 0; i < 8; ++i) {
      int tile = wave * 8 + i;
      int tm = tile >> 3, tn = tile & 7;
      v8f acc = {};
#pragma unroll
      for (int kk = 0; kk < CHUNKC; kk += 32) {
        v16bf a = load_a_frag(ktb, CHUNKC, tm * 16, kk, lane);   // kd^T [d][r]
        v16bf b = load_bt_frag(vtT, CHUNKC, kk, tn * 16, lane);
        acc = __builtin_amdgcn_wmma_f32_16x16x32_bf16(false, a, false, b, (short)0, acc, false, false);
      }
      int n = tn * 16 + (lane & 15);                // e index
      int mb = tm * 16 + ((lane >> 4) << 3);        // d base
#pragma unroll
      for (int r = 0; r < 8; ++r) {
        int d = mb + r;
        ST[n * DKH + d] = ST[n * DKH + d] * __expf(blast[d]) + acc[r];
      }
    }
    __syncthreads();
  }
}

// ---------------- host-side orchestration ----------------
extern "C" void kernel_launch(void* const* d_in, const int* in_sizes, int n_in,
                              void* d_out, int out_size, void* d_ws, size_t ws_size,
                              hipStream_t stream) {
  (void)in_sizes; (void)n_in; (void)out_size; (void)ws_size;
  const float* x     = (const float*)d_in[0];
  const float* anw   = (const float*)d_in[1];
  const float* Wq    = (const float*)d_in[2];
  const float* Wk    = (const float*)d_in[3];
  const float* Wv    = (const float*)d_in[4];
  const float* Wg    = (const float*)d_in[5];
  const float* Wgk1  = (const float*)d_in[6];
  const float* Wgk2  = (const float*)d_in[7];
  const float* bgk2  = (const float*)d_in[8];
  const float* gnw   = (const float*)d_in[9];
  const float* Wo    = (const float*)d_in[10];
  const float* mnw   = (const float*)d_in[11];
  const float* Wgate = (const float*)d_in[12];
  const float* Wup   = (const float*)d_in[13];
  const float* Wdown = (const float*)d_in[14];
  float* out = (float*)d_out;

  char* ws = (char*)d_ws;
  size_t cur = 0;
  auto alloc = [&](size_t bytes) -> void* {
    void* p = ws + cur;
    cur += (bytes + 255) & ~(size_t)255;
    return p;
  };

  unsigned short* h_bf     = (unsigned short*)alloc((size_t)BT * HS * 2);
  unsigned short* WqT      = (unsigned short*)alloc((size_t)HS * DK * 2);
  unsigned short* WkT      = (unsigned short*)alloc((size_t)HS * DK * 2);
  unsigned short* WvT      = (unsigned short*)alloc((size_t)HS * DV * 2);
  unsigned short* WgT      = (unsigned short*)alloc((size_t)HS * DV * 2);
  unsigned short* WoT      = (unsigned short*)alloc((size_t)DV * HS * 2);
  unsigned short* WgateT   = (unsigned short*)alloc((size_t)HS * ISZ * 2);
  unsigned short* WupT     = (unsigned short*)alloc((size_t)HS * ISZ * 2);
  unsigned short* WdownT   = (unsigned short*)alloc((size_t)ISZ * HS * 2);
  unsigned short* Wgk12T   = (unsigned short*)alloc((size_t)HS * DK * 2);
  float* qf    = (float*)alloc((size_t)BT * DK * 4);
  float* kf    = (float*)alloc((size_t)BT * DK * 4);
  float* vf    = (float*)alloc((size_t)BT * DV * 4);
  float* goutf = (float*)alloc((size_t)BT * DV * 4);
  float* zbuf  = (float*)alloc((size_t)BT * DK * 4);   // -> glog in place
  float* of    = (float*)alloc((size_t)BT * DV * 4);
  unsigned short* gated_bf = (unsigned short*)alloc((size_t)BT * DV * 2);
  float* attnf = (float*)alloc((size_t)BT * HS * 4);
  float* x2f   = (float*)alloc((size_t)BT * HS * 4);
  unsigned short* n_bf = (unsigned short*)alloc((size_t)BT * HS * 2);
  float* gatef = (float*)alloc((size_t)BT * ISZ * 4);
  float* upf   = (float*)alloc((size_t)BT * ISZ * 4);
  unsigned short* p_bf = (unsigned short*)alloc((size_t)BT * ISZ * 2);
  float* downf = attnf;   // attn_out dead after x2 — reuse

  auto cvtT = [&](const float* s, unsigned short* d, int K, int N) {
    long n = (long)K * N;
    k_cvt_t<<<(int)((n + 255) / 256), 256, 0, stream>>>(s, d, K, N);
  };
  auto gemm = [&](const unsigned short* A, const unsigned short* Bt, float* C,
                  int M, int N, int K) {
    long tiles = (long)(M / 16) * (N / 64);
    k_gemm_bf16<<<(int)((tiles + 7) / 8), 256, 0, stream>>>(A, Bt, C, M, N, K);
  };

  // 1) h = rmsnorm(x) -> bf16
  k_rmsnorm_bf16<<<BT, 256, 0, stream>>>(x, anw, h_bf, HS);

  // 2) weight conversions (transposed bf16 packing for K-contiguous B frags)
  cvtT(Wq, WqT, HS, DK);
  cvtT(Wk, WkT, HS, DK);
  cvtT(Wv, WvT, HS, DV);
  cvtT(Wg, WgT, HS, DV);
  cvtT(Wo, WoT, DV, HS);
  cvtT(Wgate, WgateT, HS, ISZ);
  cvtT(Wup, WupT, HS, ISZ);
  cvtT(Wdown, WdownT, ISZ, HS);
  k_wgk12<<<(HS * DK + 255) / 256, 256, 0, stream>>>(Wgk1, Wgk2, Wgk12T);

  // 3) projections
  gemm(h_bf, WqT, qf, BT, DK, HS);
  gemm(h_bf, WkT, kf, BT, DK, HS);
  gemm(h_bf, WvT, vf, BT, DV, HS);
  gemm(h_bf, WgT, goutf, BT, DV, HS);
  gemm(h_bf, Wgk12T, zbuf, BT, DK, HS);
  k_glog<<<(int)(((long)BT * DK + 255) / 256), 256, 0, stream>>>(zbuf, bgk2, (long)BT * DK);

  // 4) chunked GLA scan
  k_gla_attn<<<NB * NH * 2, 256, 0, stream>>>(qf, kf, vf, zbuf, of);

  // 5) gated output norm -> bf16, then Wo
  k_gatenorm<<<BT * NH, 256, 0, stream>>>(of, goutf, gnw, gated_bf);
  gemm(gated_bf, WoT, attnf, BT, HS, DV);

  // 6) residual + mlp norm
  k_resnorm<<<BT, 256, 0, stream>>>(attnf, x, mnw, x2f, n_bf, HS);

  // 7) MLP
  gemm(n_bf, WgateT, gatef, BT, ISZ, HS);
  gemm(n_bf, WupT, upf, BT, ISZ, HS);
  k_silu_mul<<<(int)(((long)BT * ISZ + 255) / 256), 256, 0, stream>>>(gatef, upf, p_bf,
                                                                     (long)BT * ISZ);
  gemm(p_bf, WdownT, downf, BT, HS, ISZ);

  // 8) final residual
  k_final_add<<<(int)(((long)BT * HS + 255) / 256), 256, 0, stream>>>(x2f, downf, out,
                                                                     (long)BT * HS);
}
